// BaseGINE_54752243090035
// MI455X (gfx1250) — matrix operand order repaired and checked by reading, not compiled, not run on hardware
//
#include <hip/hip_runtime.h>
#include <hip/hip_bf16.h>

#define NN 100000
#define EE 600000
#define DD 128
#define EDD 16
#define LL 4
#define NTILE (NN / 16)   // 6250 exact

typedef __attribute__((ext_vector_type(16))) __bf16 v16bf;
typedef __attribute__((ext_vector_type(8)))  __bf16 v8bf;
typedef __attribute__((ext_vector_type(8)))  float  v8f;

// Native f32 -> bf16 conversion: let the backend pick the hw converter
// (pairs fuse into v_cvt_pk_bf16_f32) instead of manual bit-twiddling VALU.
__device__ __forceinline__ __bf16 f2bf(float f) { return (__bf16)f; }

// ---------------------------------------------------------------------------
// Prepack W1/W2 (L,128,128 fp32, row-major K x N) into bf16 WMMA B-fragments.
// B 32x16 bf16 layout (wave32): lane holds column N = lane&15,
// K = (lane>>4)*8 + j for j<8, 16 + (lane>>4)*8 + (j-8) for j>=8 (matches A).
// Fragment f = kt*8+nt stored at f*512 + lane*16 halves.
// ---------------------------------------------------------------------------
__global__ void gine_prepack_kernel(const float* __restrict__ W1,
                                    const float* __restrict__ W2,
                                    __bf16* __restrict__ wpk) {
    int g = blockIdx.x * 256 + threadIdx.x;           // LL*2*32*32 = 8192
    if (g >= LL * 2 * 32 * 32) return;
    int lane = g & 31;
    int frag = (g >> 5) & 31;                         // kt*8 + nt
    int m    = (g >> 10) & 1;
    int l    = g >> 11;
    int kt = frag >> 3, nt = frag & 7;
    const float* W = (m ? W2 : W1) + l * DD * DD;
    int kb  = kt * 32 + (lane >> 4) * 8;
    int col = nt * 16 + (lane & 15);
    __bf16* dst = wpk + (size_t)(l * 2 + m) * DD * DD + frag * 512 + lane * 16;
#pragma unroll
    for (int j = 0; j < 16; ++j)
        dst[j] = f2bf(W[(kb + j + (j >= 8 ? 8 : 0)) * DD + col]);
}

// ---------------------------------------------------------------------------
// Edge phase: fused e_emb = edge_attr @ We + be (We in LDS, 8 KB), then
// msg = relu(x[src] + e_emb) * ew, scatter-add to agg via hw f32 atomics.
// 32 lanes per edge, 4 features per lane (float4 coalesced on x[src]).
// ---------------------------------------------------------------------------
__global__ __launch_bounds__(256) void gine_edge_kernel(
    const float* __restrict__ x, const int* __restrict__ ei,
    const float* __restrict__ eattr, const float* __restrict__ ew,
    const float* __restrict__ We, const float* __restrict__ be,
    float* __restrict__ agg) {
    __shared__ float Wes[EDD * DD];
    for (int i = threadIdx.x; i < EDD * DD; i += 256) Wes[i] = We[i];
    __syncthreads();

    int g = blockIdx.x * 256 + threadIdx.x;
    int e = g >> 5;
    if (e >= EE) return;
    int c4 = (g & 31) * 4;

    int src = ei[e];
    int dst = ei[EE + e];
    float w = ew[e];

    const float4* ea = (const float4*)(eattr + (size_t)e * EDD);
    float4 A0 = ea[0], A1 = ea[1], A2 = ea[2], A3 = ea[3];
    float ak[16] = {A0.x, A0.y, A0.z, A0.w, A1.x, A1.y, A1.z, A1.w,
                    A2.x, A2.y, A2.z, A2.w, A3.x, A3.y, A3.z, A3.w};

    float4 acc = *(const float4*)(be + c4);
#pragma unroll
    for (int k = 0; k < 16; ++k) {
        float4 wr = *(const float4*)&Wes[k * DD + c4];
        acc.x = fmaf(ak[k], wr.x, acc.x);
        acc.y = fmaf(ak[k], wr.y, acc.y);
        acc.z = fmaf(ak[k], wr.z, acc.z);
        acc.w = fmaf(ak[k], wr.w, acc.w);
    }

    float4 xv = *(const float4*)(x + (size_t)src * DD + c4);
    float4 m;
    m.x = fmaxf(xv.x + acc.x, 0.f) * w;
    m.y = fmaxf(xv.y + acc.y, 0.f) * w;
    m.z = fmaxf(xv.z + acc.z, 0.f) * w;
    m.w = fmaxf(xv.w + acc.w, 0.f) * w;

    float* ag = agg + (size_t)dst * DD + c4;
    unsafeAtomicAdd(ag + 0, m.x);
    unsafeAtomicAdd(ag + 1, m.y);
    unsafeAtomicAdd(ag + 2, m.z);
    unsafeAtomicAdd(ag + 3, m.w);
}

// ---------------------------------------------------------------------------
// Node phase: one wave per 16-node tile.
// hin = (1+eps)*x + agg (built directly into bf16 A-fragments)
// GEMM1: 4 K-steps x 8 N-tiles of v_wmma_f32_16x16x32_bf16, +b1, relu,
// staged to LDS as bf16, reloaded as A-fragments,
// GEMM2: 32 more WMMAs, +b2, BN, relu; final h staged to fp32 LDS so the
// residual update x += h is done with coalesced float4 ld/st.
// A 16x32 bf16 layout: row = lane&15; lanes0-15: elems 0-7 -> K kb..kb+7,
// elems 8-15 -> K kb+16..kb+23 with kb=kt*32 (+8 for lanes16-31).
// C layout: M = v + 8*(lane>>4), N = lane&15 per 16-col tile.
// ---------------------------------------------------------------------------
__global__ __launch_bounds__(128) void gine_node_kernel(
    float* __restrict__ x, const float* __restrict__ agg,
    const __bf16* __restrict__ wpk,
    const float* __restrict__ b1, const float* __restrict__ b2,
    const float* __restrict__ epsArr,
    const float* __restrict__ gamma, const float* __restrict__ beta,
    const float* __restrict__ rmean, const float* __restrict__ rvar,
    int layer) {
    __shared__ __bf16 stage[4][16 * DD];    // 16 KB: relu(h@W1+b1) in bf16
    __shared__ float  stagef[4][16 * DD];   // 32 KB: final h in fp32

    int wave = threadIdx.x >> 5;
    int lane = threadIdx.x & 31;
    int tile = blockIdx.x * 4 + wave;
    if (tile >= NTILE) return;            // no block barrier below -> safe

    int node0 = tile * 16;
    int r     = lane & 15;
    int khi   = lane >> 4;

    float epsv = 1.0f + epsArr[layer];
    const __bf16* Wp1 = wpk + (size_t)(layer * 2 + 0) * DD * DD;
    const __bf16* Wp2 = wpk + (size_t)(layer * 2 + 1) * DD * DD;
    const float* b1l = b1    + layer * DD;
    const float* b2l = b2    + layer * DD;
    const float* gl  = gamma + layer * DD;
    const float* btl = beta  + layer * DD;
    const float* ml  = rmean + layer * DD;
    const float* vl  = rvar  + layer * DD;

    const float* xr = x   + (size_t)(node0 + r) * DD;
    const float* ar = agg + (size_t)(node0 + r) * DD;

    // ---------------- GEMM 1 ----------------
    v8f c[8];
#pragma unroll
    for (int nt = 0; nt < 8; ++nt) c[nt] = (v8f)(0.0f);

#pragma unroll
    for (int kt = 0; kt < 4; ++kt) {
        int kb = kt * 32 + khi * 8;
        float4 xa = *(const float4*)(xr + kb);
        float4 xb = *(const float4*)(xr + kb + 4);
        float4 xc = *(const float4*)(xr + kb + 16);
        float4 xd = *(const float4*)(xr + kb + 20);
        float4 aa = *(const float4*)(ar + kb);
        float4 ab = *(const float4*)(ar + kb + 4);
        float4 ac = *(const float4*)(ar + kb + 16);
        float4 ad = *(const float4*)(ar + kb + 20);
        v16bf a;
        a[0]  = f2bf(fmaf(epsv, xa.x, aa.x));
        a[1]  = f2bf(fmaf(epsv, xa.y, aa.y));
        a[2]  = f2bf(fmaf(epsv, xa.z, aa.z));
        a[3]  = f2bf(fmaf(epsv, xa.w, aa.w));
        a[4]  = f2bf(fmaf(epsv, xb.x, ab.x));
        a[5]  = f2bf(fmaf(epsv, xb.y, ab.y));
        a[6]  = f2bf(fmaf(epsv, xb.z, ab.z));
        a[7]  = f2bf(fmaf(epsv, xb.w, ab.w));
        a[8]  = f2bf(fmaf(epsv, xc.x, ac.x));
        a[9]  = f2bf(fmaf(epsv, xc.y, ac.y));
        a[10] = f2bf(fmaf(epsv, xc.z, ac.z));
        a[11] = f2bf(fmaf(epsv, xc.w, ac.w));
        a[12] = f2bf(fmaf(epsv, xd.x, ad.x));
        a[13] = f2bf(fmaf(epsv, xd.y, ad.y));
        a[14] = f2bf(fmaf(epsv, xd.z, ad.z));
        a[15] = f2bf(fmaf(epsv, xd.w, ad.w));
#pragma unroll
        for (int nt = 0; nt < 8; ++nt) {
            v16bf bfr = *(const v16bf*)(Wp1 + (kt * 8 + nt) * 512 + lane * 16);
            c[nt] = __builtin_amdgcn_wmma_f32_16x16x32_bf16(
                false, a, false, bfr, (short)0, c[nt], false, false);
        }
    }

    // bias + relu, stage to LDS as bf16 (same-wave DS ops are in-order)
#pragma unroll
    for (int nt = 0; nt < 8; ++nt) {
        float bv = b1l[nt * 16 + r];
#pragma unroll
        for (int v = 0; v < 8; ++v) {
            float h = fmaxf(c[nt][v] + bv, 0.f);
            stage[wave][(v + 8 * khi) * DD + nt * 16 + r] = f2bf(h);
        }
    }

    // ---------------- GEMM 2 ----------------
    v8f c2[8];
#pragma unroll
    for (int nt = 0; nt < 8; ++nt) c2[nt] = (v8f)(0.0f);

#pragma unroll
    for (int kt = 0; kt < 4; ++kt) {
        int kb = kt * 32 + khi * 8;
        v8bf lo = *(const v8bf*)&stage[wave][r * DD + kb];
        v8bf hi = *(const v8bf*)&stage[wave][r * DD + kb + 16];
        v16bf a = __builtin_shufflevector(lo, hi, 0, 1, 2, 3, 4, 5, 6, 7,
                                          8, 9, 10, 11, 12, 13, 14, 15);
#pragma unroll
        for (int nt = 0; nt < 8; ++nt) {
            v16bf bfr = *(const v16bf*)(Wp2 + (kt * 8 + nt) * 512 + lane * 16);
            c2[nt] = __builtin_amdgcn_wmma_f32_16x16x32_bf16(
                false, a, false, bfr, (short)0, c2[nt], false, false);
        }
    }

    // bias + BN + relu -> fp32 LDS stage
#pragma unroll
    for (int nt = 0; nt < 8; ++nt) {
        int col = nt * 16 + r;
        float b2v = b2l[col];
        float sc  = gl[col] * rsqrtf(vl[col] + 1e-5f);
        float mu  = ml[col];
        float bt  = btl[col];
#pragma unroll
        for (int v = 0; v < 8; ++v) {
            float h = c2[nt][v] + b2v;
            h = (h - mu) * sc + bt;
            h = fmaxf(h, 0.f);
            stagef[wave][(v + 8 * khi) * DD + col] = h;
        }
    }

    // residual x += h, coalesced float4: iteration 'row' covers one node row,
    // 32 lanes cover its 128 columns.
#pragma unroll 4
    for (int row = 0; row < 16; ++row) {
        int colq = lane * 4;
        float* xp = x + (size_t)(node0 + row) * DD + colq;
        float4 xv = *(const float4*)xp;
        const float4 hv = *(const float4*)&stagef[wave][row * DD + colq];
        xv.x += hv.x; xv.y += hv.y; xv.z += hv.z; xv.w += hv.w;
        *(float4*)xp = xv;
    }
}

// ---------------------------------------------------------------------------
extern "C" void kernel_launch(void* const* d_in, const int* in_sizes, int n_in,
                              void* d_out, int out_size, void* d_ws, size_t ws_size,
                              hipStream_t stream) {
    const float* x_in  = (const float*)d_in[0];
    const int*   ei    = (const int*)d_in[1];
    const float* eattr = (const float*)d_in[2];
    const float* ew    = (const float*)d_in[3];
    const float* We    = (const float*)d_in[4];
    const float* be    = (const float*)d_in[5];
    const float* W1    = (const float*)d_in[6];
    const float* b1    = (const float*)d_in[7];
    const float* W2    = (const float*)d_in[8];
    const float* b2    = (const float*)d_in[9];
    const float* eps   = (const float*)d_in[10];
    const float* gamma = (const float*)d_in[11];
    const float* beta  = (const float*)d_in[12];
    const float* rmean = (const float*)d_in[13];
    const float* rvar  = (const float*)d_in[14];

    float* xcur = (float*)d_out;                        // running node features
    __bf16* wpk = (__bf16*)d_ws;                        // 256 KB packed weights
    float* agg  = (float*)((char*)d_ws + (512u << 10)); // 51.2 MB aggregate

    // x -> d_out (residual accumulator)
    hipMemcpyAsync(xcur, x_in, (size_t)NN * DD * sizeof(float),
                   hipMemcpyDeviceToDevice, stream);

    // pack weights to bf16 WMMA fragments
    gine_prepack_kernel<<<(LL * 2 * 32 * 32 + 255) / 256, 256, 0, stream>>>(
        W1, W2, wpk);

    int edge_blocks = (EE * 32 + 255) / 256;            // 75000
    int node_blocks = (NTILE + 3) / 4;                  // 1563

    for (int l = 0; l < LL; ++l) {
        hipMemsetAsync(agg, 0, (size_t)NN * DD * sizeof(float), stream);
        gine_edge_kernel<<<edge_blocks, 256, 0, stream>>>(
            xcur, ei, eattr, ew, We, be, agg);
        gine_node_kernel<<<node_blocks, 128, 0, stream>>>(
            xcur, agg, wpk, b1, b2, eps, gamma, beta, rmean, rvar, l);
    }
}
